// modal_graph_fusion_6236292514039
// MI455X (gfx1250) — matrix operand order repaired and checked by reading, not compiled, not run on hardware
//
#include <hip/hip_runtime.h>
#include <hip/hip_bf16.h>
#include <math.h>

typedef __bf16 bf16_t;
typedef __attribute__((ext_vector_type(16))) __bf16 v16bf;
typedef __attribute__((ext_vector_type(8)))  __bf16 v8bf;
typedef __attribute__((ext_vector_type(8)))  float  v8f;
typedef __attribute__((ext_vector_type(4)))  float  v4f;

#define BN     16384
#define FEATD  1024
#define FDD    256
#define MN     4
#define H1N    4
#define C1N    128
#define D1N    512     // H1*C1
#define C2N    256
#define OUTD   8
#define SLOPEV 0.2f
#define NEGBIG (-1e30f)

// ---------------- WMMA helpers (CDNA5 v_wmma_f32_16x16x32_bf16) ----------------

static __device__ __forceinline__ v8f vzero8() {
  v8f z;
#pragma unroll
  for (int i = 0; i < 8; ++i) z[i] = 0.0f;
  return z;
}

static __device__ __forceinline__ v8f wmma_bf16(v16bf a, v16bf b, v8f c) {
  // 8 args: (neg_a, A, neg_b, B, c_mod, C, reuse_a, reuse_b)
  return __builtin_amdgcn_wmma_f32_16x16x32_bf16(false, a, false, b, (short)0, c,
                                                 false, false);
}

// A-fragment (16x32 bf16): lane holds row (lane&15); K groups at khalf*8 and +16.
// p must point at &A[row*ld + ktile*32 + (lane>>4)*8].
static __device__ __forceinline__ v16bf load_a_bf16(const bf16_t* p) {
  v8bf lo = *(const v8bf*)(p);
  v8bf hi = *(const v8bf*)(p + 16);
  v16bf r;
#pragma unroll
  for (int i = 0; i < 8; ++i) { r[i] = lo[i]; r[i + 8] = hi[i]; }
  return r;
}

// Same, but source is fp32 (convert to bf16 in regs).
static __device__ __forceinline__ v16bf load_a_f32(const float* p) {
  v4f a0 = *(const v4f*)(p);
  v4f a1 = *(const v4f*)(p + 4);
  v4f a2 = *(const v4f*)(p + 16);
  v4f a3 = *(const v4f*)(p + 20);
  v16bf r;
#pragma unroll
  for (int i = 0; i < 4; ++i) {
    r[i]      = (__bf16)a0[i];
    r[i + 4]  = (__bf16)a1[i];
    r[i + 8]  = (__bf16)a2[i];
    r[i + 12] = (__bf16)a3[i];
  }
  return r;
}

// B-fragment (32x16 bf16): lane holds column (lane&15) of pre-transposed [K][N] weights;
// K rows at khalf*8 and +16. p must point at &B[(ktile*32 + (lane>>4)*8)*ld + col + (lane&15)].
static __device__ __forceinline__ v16bf load_b_bf16(const bf16_t* p, int ld) {
  v16bf r;
#pragma unroll
  for (int i = 0; i < 8; ++i) { r[i] = p[i * ld]; r[i + 8] = p[(i + 16) * ld]; }
  return r;
}

// Butterfly sum across the 16 lanes of each wave-half (xor masks 1,2,4,8 keep halves
// separate) -- matches C-fragment layout where the 16 columns live across lanes.
template <int N>
static __device__ __forceinline__ void half_reduce(float* v) {
#pragma unroll
  for (int m = 1; m < 16; m <<= 1) {
#pragma unroll
    for (int i = 0; i < N; ++i) v[i] += __shfl_xor(v[i], m, 32);
  }
}

static __device__ __forceinline__ float gelu_exact(float x) {
  return 0.5f * x * (1.0f + erff(x * 0.70710678118654752f));
}

// Async global->LDS copy of 64 contiguous bytes (4x B128), tracked by ASYNCcnt.
// ldsoff: LDS byte offset (single static __shared__ array sits at LDS base 0).
// gaddr : 64-bit global address. INST_OFFSET applies to both sides per ISA.
static __device__ __forceinline__ void async_copy64(unsigned ldsoff,
                                                    unsigned long long gaddr) {
  asm volatile(
      "global_load_async_to_lds_b128 %0, %1, off\n\t"
      "global_load_async_to_lds_b128 %0, %1, off offset:16\n\t"
      "global_load_async_to_lds_b128 %0, %1, off offset:32\n\t"
      "global_load_async_to_lds_b128 %0, %1, off offset:48"
      :: "v"(ldsoff), "v"(gaddr) : "memory");
}

static __device__ __forceinline__ void wait_async0() {
  asm volatile("s_wait_asynccnt 0x0" ::: "memory");
}

// ---------------- Kernel 0: weight transpose + cvt to bf16 [K][N] ----------------

#define PW (FEATD * FDD)          // 262144 per modality
#define SZ_G1 (FDD * D1N)         // 131072
#define SZ_G2 (D1N * C2N)         // 131072
#define SZ_W1 (FDD * FDD)         // 65536
#define PREP_TOTAL (4 * PW + SZ_G1 + SZ_G2 + SZ_W1)

__global__ __launch_bounds__(256) void prep_kernel(
    const float* __restrict__ Wl, const float* __restrict__ Wv,
    const float* __restrict__ Wa, const float* __restrict__ Wi,
    const float* __restrict__ g1W, const float* __restrict__ g2W,
    const float* __restrict__ hW1,
    bf16_t* __restrict__ wtp, bf16_t* __restrict__ g1t,
    bf16_t* __restrict__ g2t, bf16_t* __restrict__ w1t) {
  int idx = blockIdx.x * blockDim.x + threadIdx.x;
  if (idx >= PREP_TOTAL) return;
  if (idx < 4 * PW) {
    int m = idx / PW, r = idx % PW;
    int k = r / FDD, n = r % FDD;                 // out: [1024][256]
    const float* W = (m == 0) ? Wl : (m == 1) ? Wv : (m == 2) ? Wa : Wi;
    wtp[idx] = (bf16_t)W[n * FEATD + k];          // W is [256][1024]
    return;
  }
  idx -= 4 * PW;
  if (idx < SZ_G1) {
    int k = idx / D1N, n = idx % D1N;             // out: [256][512]
    g1t[idx] = (bf16_t)g1W[n * FDD + k];          // g1_W is [512][256]
    return;
  }
  idx -= SZ_G1;
  if (idx < SZ_G2) {
    int k = idx / C2N, n = idx % C2N;             // out: [512][256]
    g2t[idx] = (bf16_t)g2W[n * D1N + k];          // g2_W is [256][512]
    return;
  }
  idx -= SZ_G2;
  {
    int k = idx / FDD, n = idx % FDD;             // out: [256][256]
    w1t[idx] = (bf16_t)hW1[n * FDD + k];
  }
}

// ---------------- Kernel 1: modality projections (fp32 in -> bf16 feats) --------
// grid = (256, 4), block = 256 (8 waves). Block: 64 rows x 256 cols; weight slabs
// staged to LDS with GLOBAL_LOAD_ASYNC_TO_LDS_B128 (ASYNCcnt).

__global__ __launch_bounds__(256) void proj_kernel(
    const float* __restrict__ lang, const float* __restrict__ vid,
    const float* __restrict__ aud, const float* __restrict__ img,
    const float* __restrict__ bl, const float* __restrict__ bv,
    const float* __restrict__ ba, const float* __restrict__ bi,
    const bf16_t* __restrict__ wtp, bf16_t* __restrict__ feats) {
  __shared__ __align__(32) bf16_t smem[32 * FDD];   // 16KB weight slab (LDS base 0)

  const int mod = blockIdx.y;
  const float* X    = (mod == 0) ? lang : (mod == 1) ? vid : (mod == 2) ? aud : img;
  const float* bias = (mod == 0) ? bl   : (mod == 1) ? bv  : (mod == 2) ? ba  : bi;
  const bf16_t* Wt  = wtp + (size_t)mod * FEATD * FDD;

  const int tid  = threadIdx.x;
  const int lane = tid & 31;
  const int wave = tid >> 5;
  const int ln   = lane & 15;
  const int half = lane >> 4;
  const int rb   = blockIdx.x * 64 + (wave >> 1) * 16;
  const int cb   = (wave & 1) * 128;

  v8f acc[8];
#pragma unroll
  for (int t = 0; t < 8; ++t) acc[t] = vzero8();

  const float* aptr = X + (size_t)(rb + ln) * FEATD + half * 8;

  // per-thread async slab copy indexing: 64B per thread, 256 threads = 16KB tile
  const int cr = tid >> 3;            // 0..31 (K row within slab)
  const int cc = (tid & 7) * 32;      // column start (elements)
  const unsigned ldsoff = (unsigned)((cr * FDD + cc) * sizeof(bf16_t));

  for (int kt = 0; kt < FEATD / 32; ++kt) {
    async_copy64(ldsoff,
                 (unsigned long long)(Wt + (size_t)(kt * 32 + cr) * FDD + cc));
    if (kt + 1 < FEATD / 32)      // prefetch next slab (global_prefetch_b8)
      __builtin_prefetch(Wt + (size_t)(kt + 1) * 32 * FDD + tid * 32, 0, 1);
    wait_async0();
    __syncthreads();
    v16bf a = load_a_f32(aptr + kt * 32);
#pragma unroll
    for (int t = 0; t < 8; ++t) {
      v16bf b = load_b_bf16(smem + half * 8 * FDD + cb + t * 16 + ln, FDD);
      acc[t] = wmma_bf16(a, b, acc[t]);
    }
    __syncthreads();
  }

#pragma unroll
  for (int t = 0; t < 8; ++t) {
    const int n  = cb + t * 16 + ln;
    const float bb = bias[n];
#pragma unroll
    for (int r = 0; r < 8; ++r) {
      const int brow = rb + half * 8 + r;           // = sample b
      feats[((size_t)brow * MN + mod) * FDD + n] = (bf16_t)(acc[t][r] + bb);
    }
  }
}

// ---------------- Kernel 2: gat1 (lin + SuperGAT attn + GELU) -------------------
// grid = 4096, block = 128 (4 waves). Block = 16 rows (4 samples); wave = head.

__global__ __launch_bounds__(128) void gat1_kernel(
    const bf16_t* __restrict__ feats, const bf16_t* __restrict__ g1t,
    const float* __restrict__ att_l, const float* __restrict__ att_r,
    const float* __restrict__ g1b, const int* __restrict__ missing,
    bf16_t* __restrict__ x1) {
  const int tid  = threadIdx.x;
  const int lane = tid & 31;
  const int head = tid >> 5;
  const int ln   = lane & 15;
  const int half = lane >> 4;
  const int rb   = blockIdx.x * 16;

  v8f acc[8];
#pragma unroll
  for (int t = 0; t < 8; ++t) acc[t] = vzero8();

  const bf16_t* aptr = feats + (size_t)(rb + ln) * FDD + half * 8;
  for (int kt = 0; kt < FDD / 32; ++kt) {
    v16bf a = load_a_bf16(aptr + kt * 32);
#pragma unroll
    for (int t = 0; t < 8; ++t) {
      v16bf b = load_b_bf16(g1t + (size_t)(kt * 32 + half * 8) * D1N + head * C1N + t * 16 + ln, D1N);
      acc[t] = wmma_bf16(a, b, acc[t]);
    }
  }

  float h[8][8];
#pragma unroll
  for (int t = 0; t < 8; ++t)
#pragma unroll
    for (int r = 0; r < 8; ++r) h[t][r] = acc[t][r];

  float attl[8], attr[8];
#pragma unroll
  for (int t = 0; t < 8; ++t) {
    attl[t] = att_l[head * C1N + t * 16 + ln];
    attr[t] = att_r[head * C1N + t * 16 + ln];
  }

#pragma unroll
  for (int ss = 0; ss < 2; ++ss) {
    const int samp = blockIdx.x * 4 + half * 2 + ss;
    float red[24];
#pragma unroll
    for (int j = 0; j < 4; ++j) {
      float al = 0.f, ar = 0.f;
#pragma unroll
      for (int t = 0; t < 8; ++t) {
        al += h[t][ss * 4 + j] * attl[t];
        ar += h[t][ss * 4 + j] * attr[t];
      }
      red[j] = al; red[4 + j] = ar;
    }
#pragma unroll
    for (int i = 0; i < 4; ++i)
#pragma unroll
      for (int j = 0; j < 4; ++j) {
        float lg = 0.f;
#pragma unroll
        for (int t = 0; t < 8; ++t) lg += h[t][ss * 4 + i] * h[t][ss * 4 + j];
        red[8 + i * 4 + j] = lg;
      }
    half_reduce<24>(red);

    const int mi = missing[samp];
    bool pres[4];
#pragma unroll
    for (int m = 0; m < 4; ++m) pres[m] = (mi != m + 1);

    float attn[4][4];
#pragma unroll
    for (int i = 0; i < 4; ++i) {
      float av[4];
#pragma unroll
      for (int j = 0; j < 4; ++j) {
        bool ok = (i == j) || (pres[i] && pres[j]);
        float a = red[4 + i] + red[j];
        float sg = 1.0f / (1.0f + expf(-red[8 + i * 4 + j]));
        float v = a * sg;
        v = (v >= 0.f) ? v : SLOPEV * v;
        av[j] = ok ? v : NEGBIG;
      }
      float mx = fmaxf(fmaxf(av[0], av[1]), fmaxf(av[2], av[3]));
      float ex[4], sm = 0.f;
#pragma unroll
      for (int j = 0; j < 4; ++j) { ex[j] = expf(av[j] - mx); sm += ex[j]; }
      float inv = 1.0f / sm;
#pragma unroll
      for (int j = 0; j < 4; ++j) attn[i][j] = ex[j] * inv;
    }

#pragma unroll
    for (int t = 0; t < 8; ++t) {
      const int ch = head * C1N + t * 16 + ln;
      const float bb = g1b[ch];
#pragma unroll
      for (int i = 0; i < 4; ++i) {
        float o = 0.f;
#pragma unroll
        for (int j = 0; j < 4; ++j) o += attn[i][j] * h[t][ss * 4 + j];
        o = gelu_exact(o + bb);
        x1[(size_t)(rb + half * 8 + ss * 4 + i) * D1N + ch] = (bf16_t)o;
      }
    }
  }
}

// ---------------- Kernel 3: gat2 + node-mean pool + LayerNorm -------------------
// grid = 1024, block = 128 (4 waves); wave = 16 rows x 256 cols.

__global__ __launch_bounds__(128) void gat2_kernel(
    const bf16_t* __restrict__ x1, const bf16_t* __restrict__ g2t,
    const float* __restrict__ att_l, const float* __restrict__ att_r,
    const float* __restrict__ g2b, const float* __restrict__ lng,
    const float* __restrict__ lnb, const int* __restrict__ missing,
    bf16_t* __restrict__ normed) {
  const int tid  = threadIdx.x;
  const int lane = tid & 31;
  const int wave = tid >> 5;
  const int ln   = lane & 15;
  const int half = lane >> 4;
  const int rb   = (blockIdx.x * 4 + wave) * 16;

  v8f acc[16];
#pragma unroll
  for (int t = 0; t < 16; ++t) acc[t] = vzero8();

  const bf16_t* aptr = x1 + (size_t)(rb + ln) * D1N + half * 8;
  for (int kt = 0; kt < D1N / 32; ++kt) {
    v16bf a = load_a_bf16(aptr + kt * 32);
#pragma unroll
    for (int t = 0; t < 16; ++t) {
      v16bf b = load_b_bf16(g2t + (size_t)(kt * 32 + half * 8) * C2N + t * 16 + ln, C2N);
      acc[t] = wmma_bf16(a, b, acc[t]);
    }
  }

  float h[16][8];
#pragma unroll
  for (int t = 0; t < 16; ++t)
#pragma unroll
    for (int r = 0; r < 8; ++r) h[t][r] = acc[t][r];

  float attl[16], attr[16];
#pragma unroll
  for (int t = 0; t < 16; ++t) {
    attl[t] = att_l[t * 16 + ln];     // H2 == 1
    attr[t] = att_r[t * 16 + ln];
  }

#pragma unroll
  for (int ss = 0; ss < 2; ++ss) {
    const int samp = rb / 4 + half * 2 + ss;
    float red[24];
#pragma unroll
    for (int j = 0; j < 4; ++j) {
      float al = 0.f, ar = 0.f;
#pragma unroll
      for (int t = 0; t < 16; ++t) {
        al += h[t][ss * 4 + j] * attl[t];
        ar += h[t][ss * 4 + j] * attr[t];
      }
      red[j] = al; red[4 + j] = ar;
    }
#pragma unroll
    for (int i = 0; i < 4; ++i)
#pragma unroll
      for (int j = 0; j < 4; ++j) {
        float lg = 0.f;
#pragma unroll
        for (int t = 0; t < 16; ++t) lg += h[t][ss * 4 + i] * h[t][ss * 4 + j];
        red[8 + i * 4 + j] = lg;
      }
    half_reduce<24>(red);

    const int mi = missing[samp];
    bool pres[4];
#pragma unroll
    for (int m = 0; m < 4; ++m) pres[m] = (mi != m + 1);

    float attn[4][4];
#pragma unroll
    for (int i = 0; i < 4; ++i) {
      float av[4];
#pragma unroll
      for (int j = 0; j < 4; ++j) {
        bool ok = (i == j) || (pres[i] && pres[j]);
        float a = red[4 + i] + red[j];
        float sg = 1.0f / (1.0f + expf(-red[8 + i * 4 + j]));
        float v = a * sg;
        v = (v >= 0.f) ? v : SLOPEV * v;
        av[j] = ok ? v : NEGBIG;
      }
      float mx = fmaxf(fmaxf(av[0], av[1]), fmaxf(av[2], av[3]));
      float ex[4], sm = 0.f;
#pragma unroll
      for (int j = 0; j < 4; ++j) { ex[j] = expf(av[j] - mx); sm += ex[j]; }
      float inv = 1.0f / sm;
#pragma unroll
      for (int j = 0; j < 4; ++j) attn[i][j] = ex[j] * inv;
    }

    // pooled over the 4 nodes (+bias), then LayerNorm over 256 channels
    float pooled[16];
#pragma unroll
    for (int t = 0; t < 16; ++t) {
      float p = 0.f;
#pragma unroll
      for (int i = 0; i < 4; ++i) {
        float o = 0.f;
#pragma unroll
        for (int j = 0; j < 4; ++j) o += attn[i][j] * h[t][ss * 4 + j];
        p += o;
      }
      pooled[t] = p * 0.25f + g2b[t * 16 + ln];
    }
    float mu_a[1] = {0.f};
#pragma unroll
    for (int t = 0; t < 16; ++t) mu_a[0] += pooled[t];
    half_reduce<1>(mu_a);
    const float mu = mu_a[0] * (1.0f / 256.0f);
    float va[1] = {0.f};
#pragma unroll
    for (int t = 0; t < 16; ++t) { float d = pooled[t] - mu; va[0] += d * d; }
    half_reduce<1>(va);
    const float rinv = rsqrtf(va[0] * (1.0f / 256.0f) + 1e-5f);
#pragma unroll
    for (int t = 0; t < 16; ++t) {
      const int ch = t * 16 + ln;
      float nv = (pooled[t] - mu) * rinv * lng[ch] + lnb[ch];
      normed[(size_t)samp * C2N + ch] = (bf16_t)nv;
    }
  }
}

// ---------------- Kernel 4: head MLP (WMMA + ReLU + 256->8 matvec) --------------
// grid = 256, block = 128 (4 waves); wave = 16 rows x 256 cols.

__global__ __launch_bounds__(128) void head_kernel(
    const bf16_t* __restrict__ normed, const bf16_t* __restrict__ w1t,
    const float* __restrict__ b1, const float* __restrict__ W2,
    const float* __restrict__ b2, float* __restrict__ out) {
  const int tid  = threadIdx.x;
  const int lane = tid & 31;
  const int wave = tid >> 5;
  const int ln   = lane & 15;
  const int half = lane >> 4;
  const int rb   = (blockIdx.x * 4 + wave) * 16;

  v8f acc[16];
#pragma unroll
  for (int t = 0; t < 16; ++t) acc[t] = vzero8();

  const bf16_t* aptr = normed + (size_t)(rb + ln) * FDD + half * 8;
  for (int kt = 0; kt < FDD / 32; ++kt) {
    v16bf a = load_a_bf16(aptr + kt * 32);
#pragma unroll
    for (int t = 0; t < 16; ++t) {
      v16bf b = load_b_bf16(w1t + (size_t)(kt * 32 + half * 8) * FDD + t * 16 + ln, FDD);
      acc[t] = wmma_bf16(a, b, acc[t]);
    }
  }

  float hid[16][8];
#pragma unroll
  for (int t = 0; t < 16; ++t) {
    const float bb = b1[t * 16 + ln];
#pragma unroll
    for (int r = 0; r < 8; ++r) {
      float v = acc[t][r] + bb;
      hid[t][r] = (v > 0.f) ? v : 0.f;
    }
  }

  float red[64];
#pragma unroll
  for (int i = 0; i < 64; ++i) red[i] = 0.f;
#pragma unroll
  for (int o = 0; o < 8; ++o)
#pragma unroll
    for (int t = 0; t < 16; ++t) {
      const float w = W2[o * FDD + t * 16 + ln];
#pragma unroll
      for (int r = 0; r < 8; ++r) red[r * 8 + o] += hid[t][r] * w;
    }
  half_reduce<64>(red);

#pragma unroll
  for (int r = 0; r < 8; ++r)
#pragma unroll
    for (int o = 0; o < 8; ++o)
      if (ln == o)
        out[(size_t)(rb + half * 8 + r) * OUTD + o] = red[r * 8 + o] + b2[o];
}

// ---------------- Host launch ---------------------------------------------------

extern "C" void kernel_launch(void* const* d_in, const int* in_sizes, int n_in,
                              void* d_out, int out_size, void* d_ws, size_t ws_size,
                              hipStream_t stream) {
  (void)in_sizes; (void)n_in; (void)out_size; (void)ws_size;

  const float* lang = (const float*)d_in[0];
  const float* vid  = (const float*)d_in[1];
  const float* aud  = (const float*)d_in[2];
  const float* img  = (const float*)d_in[3];
  const int*   mis  = (const int*)d_in[4];
  const float* Wl = (const float*)d_in[5],  *bl = (const float*)d_in[6];
  const float* Wv = (const float*)d_in[7],  *bv = (const float*)d_in[8];
  const float* Wa = (const float*)d_in[9],  *ba = (const float*)d_in[10];
  const float* Wi = (const float*)d_in[11], *bi = (const float*)d_in[12];
  const float* g1W = (const float*)d_in[13];
  const float* g1al = (const float*)d_in[14], *g1ar = (const float*)d_in[15];
  const float* g1b  = (const float*)d_in[16];
  const float* g2W = (const float*)d_in[17];
  const float* g2al = (const float*)d_in[18], *g2ar = (const float*)d_in[19];
  const float* g2b  = (const float*)d_in[20];
  const float* lng = (const float*)d_in[21], *lnb = (const float*)d_in[22];
  const float* hW1 = (const float*)d_in[23], *hb1 = (const float*)d_in[24];
  const float* hW2 = (const float*)d_in[25], *hb2 = (const float*)d_in[26];
  float* outp = (float*)d_out;

  char* ws = (char*)d_ws;
  bf16_t* wtp   = (bf16_t*)(ws + 0);           // 4x[1024][256]  2 MB
  bf16_t* g1t   = (bf16_t*)(ws + 2097152);     // [256][512]     256 KB
  bf16_t* g2t   = (bf16_t*)(ws + 2359296);     // [512][256]     256 KB
  bf16_t* w1t   = (bf16_t*)(ws + 2621440);     // [256][256]     128 KB
  bf16_t* feats = (bf16_t*)(ws + 2752512);     // [65536][256]   32 MB
  bf16_t* x1    = (bf16_t*)(ws + 36306944);    // [65536][512]   64 MB
  bf16_t* nrm   = (bf16_t*)(ws + 103415808);   // [16384][256]   8 MB

  prep_kernel<<<(PREP_TOTAL + 255) / 256, 256, 0, stream>>>(
      Wl, Wv, Wa, Wi, g1W, g2W, hW1, wtp, g1t, g2t, w1t);

  proj_kernel<<<dim3(BN / 64, 4), 256, 0, stream>>>(
      lang, vid, aud, img, bl, bv, ba, bi, wtp, feats);

  gat1_kernel<<<(BN * MN) / 16, 128, 0, stream>>>(
      feats, g1t, g1al, g1ar, g1b, mis, x1);

  gat2_kernel<<<(BN * MN) / 64, 128, 0, stream>>>(
      x1, g2t, g2al, g2ar, g2b, lng, lnb, mis, nrm);

  head_kernel<<<BN / 64, 128, 0, stream>>>(
      nrm, w1t, hb1, hW2, hb2, outp);
}
